// Attention_49709951484392
// MI455X (gfx1250) — matrix operand order repaired and checked by reading, not compiled, not run on hardware
//
#include <hip/hip_runtime.h>
#include <hip/hip_bf16.h>

// ---------------------------------------------------------------------------
// Transformer block (LN1 -> QKV -> smeared-key ALiBi causal attention -> out
// proj -> LN2) for MI455X / gfx1250.  GEMMs run on v_wmma_f32_16x16x32_bf16
// with the weight panel staged through LDS via async global->LDS copies
// (ASYNCcnt double buffering), last iteration peeled for branch-free steady
// state.
// ---------------------------------------------------------------------------

typedef __bf16 bf16_t;
typedef __attribute__((ext_vector_type(16))) __bf16 v16bf;
typedef __attribute__((ext_vector_type(8)))  __bf16 v8bf;
typedef __attribute__((ext_vector_type(8)))  float  v8f;
typedef int a4i __attribute__((vector_size(16)));   // builtin's expected pointee

#define HEADS   16
#define DHEAD   64
#define DMODEL  1024
#define BATCH   2
#define SEQ     2048
#define ROWS    (BATCH * SEQ)          // 4096
#define QKVCOLS (3 * DMODEL)           // 3072

static __device__ __forceinline__ v16bf cat8(v8bf lo, v8bf hi) {
  return __builtin_shufflevector(lo, hi, 0,1,2,3,4,5,6,7,8,9,10,11,12,13,14,15);
}

static __device__ __forceinline__ v8f wmma_bf16(v16bf a, v16bf b, v8f c) {
  // 8 args: (neg_a, A, neg_b, B, c_mod, C, reuse_a, reuse_b)
  return __builtin_amdgcn_wmma_f32_16x16x32_bf16(false, a, false, b, (short)0, c,
                                                 false, false);
}

// --- async global -> LDS (16B per lane), tracked with ASYNCcnt -------------
static __device__ __forceinline__ void async_ld16(const void* g, void* l) {
#if __has_builtin(__builtin_amdgcn_global_load_async_to_lds_b128)
  __builtin_amdgcn_global_load_async_to_lds_b128(
      (__attribute__((address_space(1))) a4i*)(a4i*)(g),
      (__attribute__((address_space(3))) a4i*)(a4i*)(l), 0, 0);
#else
  unsigned lofs = (unsigned)(size_t)(__attribute__((address_space(3))) void*)(l);
  asm volatile("global_load_async_to_lds_b128 %0, %1, off"
               :: "v"(lofs), "v"(g) : "memory");
#endif
}

#if __has_builtin(__builtin_amdgcn_s_wait_asynccnt)
#define WAIT_ASYNC(n) __builtin_amdgcn_s_wait_asynccnt(n)
#else
#define WAIT_ASYNC(n) asm volatile("s_wait_asynccnt %0" :: "i"(n) : "memory")
#endif

// Reduce across a 16-lane half of the wave32 (xor masks 1..8 stay in group).
static __device__ __forceinline__ float halfmax(float v) {
  #pragma unroll
  for (int m = 1; m <= 8; m <<= 1) v = fmaxf(v, __shfl_xor(v, m, 32));
  return v;
}
static __device__ __forceinline__ float halfsum(float v) {
  #pragma unroll
  for (int m = 1; m <= 8; m <<= 1) v += __shfl_xor(v, m, 32);
  return v;
}

// ---------------------------------------------------------------------------
// fp32 -> bf16 conversion (weights)
// ---------------------------------------------------------------------------
__global__ __launch_bounds__(256) void k_f2bf(const float* __restrict__ src,
                                              bf16_t* __restrict__ dst, int n) {
  int i = blockIdx.x * 256 + threadIdx.x;
  if (i < n) dst[i] = (bf16_t)src[i];
}

// ---------------------------------------------------------------------------
// LN1: one block per row of x, output bf16 activations
// ---------------------------------------------------------------------------
__global__ __launch_bounds__(256) void k_ln1(const float* __restrict__ x,
                                             const float* __restrict__ g,
                                             const float* __restrict__ b,
                                             bf16_t* __restrict__ hb) {
  __shared__ float s1[256], s2[256];
  int row = blockIdx.x, t = threadIdx.x;
  const float* xr = x + (size_t)row * DMODEL;
  float v[4], s = 0.f, ss = 0.f;
  #pragma unroll
  for (int i = 0; i < 4; i++) {
    float u = xr[t + 256 * i];
    v[i] = u; s += u; ss += u * u;
  }
  s1[t] = s; s2[t] = ss; __syncthreads();
  for (int off = 128; off > 0; off >>= 1) {
    if (t < off) { s1[t] += s1[t + off]; s2[t] += s2[t + off]; }
    __syncthreads();
  }
  float mu  = s1[0] * (1.0f / DMODEL);
  float var = s2[0] * (1.0f / DMODEL) - mu * mu;
  float rstd = rsqrtf(var + 1e-5f);
  #pragma unroll
  for (int i = 0; i < 4; i++) {
    int c = t + 256 * i;
    hb[(size_t)row * DMODEL + c] = (bf16_t)((v[i] - mu) * rstd * g[c] + b[c]);
  }
}

// ---------------------------------------------------------------------------
// QKV GEMM, LDS-staged weights: workgroup = 128 rows x 64 cols; all 8 waves
// share one 64x32 bf16 B panel per K-step, async-copied to LDS and
// double-buffered on ASYNCcnt.  Epilogue scatters into head-major Q (scaled
// by 1/sqrt(64)), raw K, and transposed V [bh][d][l].
// ---------------------------------------------------------------------------
__global__ __launch_bounds__(256) void k_qkv(const bf16_t* __restrict__ hb,
                                             const bf16_t* __restrict__ wib,
                                             bf16_t* __restrict__ qb,
                                             bf16_t* __restrict__ kraw,
                                             bf16_t* __restrict__ vbT) {
  __shared__ bf16_t Bs[2][64 * 32];
  int warp = threadIdx.x >> 5, lane = threadIdx.x & 31;
  int t = threadIdx.x;
  int mb  = blockIdx.x & 31;           // 32 row blocks of 128
  int nt4 = blockIdx.x >> 5;           // 48 col tiles of 64
  int m0 = (mb * 8 + warp) * 16, j0 = nt4 * 64;
  int lm = lane & 15, lh = lane >> 4;

  // staging assignment: thread t copies 16B: row t>>2, 8-elem segment t&3
  int srow = t >> 2, sseg = (t & 3) << 3;
  const bf16_t* wrow = wib + (size_t)(j0 + srow) * DMODEL + sseg;
  bf16_t* lrow = &Bs[0][srow * 32 + sseg];

  async_ld16(wrow, lrow);                          // kk = 0 into buf 0
  v8f acc[4] = {};
  const bf16_t* abase = hb + (size_t)(m0 + lm) * DMODEL + (lh << 3);
  int cur = 0;
  #pragma unroll 1
  for (int kk = 0; kk < DMODEL - 32; kk += 32) {   // steady state, branch-free
    v16bf a = cat8(*(const v8bf*)(abase + kk), *(const v8bf*)(abase + kk + 16));
    async_ld16(wrow + kk + 32, lrow + (cur ^ 1) * (64 * 32));
    WAIT_ASYNC(1);
    __syncthreads();                               // panel visible to all waves
    v16bf bfr[4];
    #pragma unroll
    for (int nt = 0; nt < 4; nt++) {
      const bf16_t* bp = &Bs[cur][(nt * 16 + lm) * 32 + (lh << 3)];
      bfr[nt] = cat8(*(const v8bf*)bp, *(const v8bf*)(bp + 16));
    }
    #pragma unroll
    for (int nt = 0; nt < 4; nt++) acc[nt] = wmma_bf16(a, bfr[nt], acc[nt]);
    __syncthreads();                               // done reading before refill
    cur ^= 1;
  }
  {                                                // peeled final iteration
    const int kk = DMODEL - 32;
    v16bf a = cat8(*(const v8bf*)(abase + kk), *(const v8bf*)(abase + kk + 16));
    WAIT_ASYNC(0);
    __syncthreads();
    v16bf bfr[4];
    #pragma unroll
    for (int nt = 0; nt < 4; nt++) {
      const bf16_t* bp = &Bs[cur][(nt * 16 + lm) * 32 + (lh << 3)];
      bfr[nt] = cat8(*(const v8bf*)bp, *(const v8bf*)(bp + 16));
    }
    #pragma unroll
    for (int nt = 0; nt < 4; nt++) acc[nt] = wmma_bf16(a, bfr[nt], acc[nt]);
  }
  #pragma unroll
  for (int nt = 0; nt < 4; nt++) {
    #pragma unroll
    for (int r = 0; r < 8; r++) {
      float val = acc[nt][r];
      int i = m0 + r + (lh << 3);
      int j = j0 + nt * 16 + lm;
      int bidx = i >> 11, l = i & (SEQ - 1);
      int sec = j >> 10, hj = (j >> 6) & 15, dd = j & 63;
      size_t hrow = ((size_t)bidx * HEADS + hj) * SEQ + l;
      if (sec == 0)      qb  [hrow * DHEAD + dd] = (bf16_t)(val * 0.125f);
      else if (sec == 1) kraw[hrow * DHEAD + dd] = (bf16_t)val;
      else vbT[(((size_t)bidx * HEADS + hj) * DHEAD + dd) * SEQ + l] = (bf16_t)val;
    }
  }
}

// ---------------------------------------------------------------------------
// Smeared keys: k[l] = (1-s)*k[l] + s*k[l-1],  s = sigmoid(smear[h])
// ---------------------------------------------------------------------------
__global__ __launch_bounds__(256) void k_smear(const bf16_t* __restrict__ kraw,
                                               const float* __restrict__ smear,
                                               bf16_t* __restrict__ kb) {
  int idx = blockIdx.x * 256 + threadIdx.x;    // BATCH*HEADS*SEQ*DHEAD
  int l  = (idx >> 6) & (SEQ - 1);
  int h  = (idx >> 17) & 15;
  float s = 1.f / (1.f + __expf(-smear[h]));
  float kc = (float)kraw[idx];
  float kp = (l > 0) ? (float)kraw[idx - DHEAD] : 0.f;
  kb[idx] = (bf16_t)((1.f - s) * kc + s * kp);
}

// ---------------------------------------------------------------------------
// Flash attention: one wave = 16 query rows of one (b,h).
// S = Q K^T via 2x2 WMMA per 32-key block, online softmax with ALiBi +
// causal mask, P staged through LDS (D-layout -> A-layout), PV via 4 WMMA.
// ---------------------------------------------------------------------------
__global__ __launch_bounds__(256) void k_attn(const bf16_t* __restrict__ qb,
                                              const bf16_t* __restrict__ kb,
                                              const bf16_t* __restrict__ vbT,
                                              const float* __restrict__ slopes,
                                              bf16_t* __restrict__ ob) {
  __shared__ bf16_t plds[8 * 16 * 32];        // per-wave 16x32 P tile
  int warp = threadIdx.x >> 5, lane = threadIdx.x & 31;
  int job = blockIdx.x * 8 + warp;
  int bh = job >> 7;                          // 32 (b,h) pairs
  int qt = job & 127;                         // 128 q-tiles of 16
  int bb = bh >> 4, h = bh & 15;
  int q0 = qt * 16;
  int lm = lane & 15, lh = lane >> 4;

  const bf16_t* qbh = qb  + (size_t)bh * SEQ * DHEAD;
  const bf16_t* kbh = kb  + (size_t)bh * SEQ * DHEAD;
  const bf16_t* vbh = vbT + (size_t)bh * DHEAD * SEQ;
  float slope = slopes[h];

  const bf16_t* qrow = qbh + (size_t)(q0 + lm) * DHEAD + (lh << 3);
  v16bf aq0 = cat8(*(const v8bf*)(qrow),      *(const v8bf*)(qrow + 16));
  v16bf aq1 = cat8(*(const v8bf*)(qrow + 32), *(const v8bf*)(qrow + 48));

  v8f accO[4] = {};
  float mrun[8], lrun[8];
  #pragma unroll
  for (int r = 0; r < 8; r++) { mrun[r] = -3e38f; lrun[r] = 0.f; }
  bf16_t* pw = plds + warp * 512;

  for (int j0 = 0; j0 <= q0 + 15; j0 += 32) {
    v8f s[2];
    #pragma unroll
    for (int t = 0; t < 2; t++) {
      int jt = j0 + t * 16;
      const bf16_t* krow = kbh + (size_t)(jt + lm) * DHEAD + (lh << 3);
      v16bf b0 = cat8(*(const v8bf*)(krow),      *(const v8bf*)(krow + 16));
      v16bf b1 = cat8(*(const v8bf*)(krow + 32), *(const v8bf*)(krow + 48));
      v8f z = {};
      z = wmma_bf16(aq0, b0, z);
      z = wmma_bf16(aq1, b1, z);
      s[t] = z;
    }
    #pragma unroll
    for (int r = 0; r < 8; r++) {
      int q = q0 + r + (lh << 3);
      float m = -3e38f;
      #pragma unroll
      for (int t = 0; t < 2; t++) {
        int j = j0 + t * 16 + lm;
        float v = s[t][r] + slope * (float)j;
        if (j > q) v = -3e38f;                // causal mask
        s[t][r] = v;
        m = fmaxf(m, v);
      }
      float rmax = halfmax(m);
      float nm = fmaxf(mrun[r], rmax);
      float sc = __expf(mrun[r] - nm);
      mrun[r] = nm;
      lrun[r] *= sc;
      #pragma unroll
      for (int nt = 0; nt < 4; nt++) accO[nt][r] *= sc;
      float psum = 0.f;
      #pragma unroll
      for (int t = 0; t < 2; t++) {
        float p = __expf(s[t][r] - nm);
        psum += p;
        pw[(r + (lh << 3)) * 32 + t * 16 + lm] = (bf16_t)p;
      }
      lrun[r] += halfsum(psum);
    }
    asm volatile("s_wait_dscnt 0" ::: "memory");   // P tile visible in LDS
    const bf16_t* prow = pw + lm * 32 + (lh << 3);
    v16bf pa = cat8(*(const v8bf*)(prow), *(const v8bf*)(prow + 16));
    #pragma unroll
    for (int nt = 0; nt < 4; nt++) {
      const bf16_t* vrow = vbh + (size_t)(nt * 16 + lm) * SEQ + j0 + (lh << 3);
      v16bf bv = cat8(*(const v8bf*)(vrow), *(const v8bf*)(vrow + 16));
      accO[nt] = wmma_bf16(pa, bv, accO[nt]);
    }
  }
  #pragma unroll
  for (int nt = 0; nt < 4; nt++) {
    #pragma unroll
    for (int r = 0; r < 8; r++) {
      int q = q0 + r + (lh << 3);
      float val = accO[nt][r] / lrun[r];
      ob[((size_t)(bb * SEQ + q)) * DMODEL + h * DHEAD + nt * 16 + lm] = (bf16_t)val;
    }
  }
}

// ---------------------------------------------------------------------------
// Output projection (LDS-staged weights, same scheme as k_qkv), fp32 out
// ---------------------------------------------------------------------------
__global__ __launch_bounds__(256) void k_proj(const bf16_t* __restrict__ obm,
                                              const bf16_t* __restrict__ wob,
                                              float* __restrict__ out) {
  __shared__ bf16_t Bs[2][64 * 32];
  int warp = threadIdx.x >> 5, lane = threadIdx.x & 31;
  int t = threadIdx.x;
  int mb  = blockIdx.x & 31;           // 32 row blocks of 128
  int nt4 = blockIdx.x >> 5;           // 16 col tiles of 64
  int m0 = (mb * 8 + warp) * 16, j0 = nt4 * 64;
  int lm = lane & 15, lh = lane >> 4;

  int srow = t >> 2, sseg = (t & 3) << 3;
  const bf16_t* wrow = wob + (size_t)(j0 + srow) * DMODEL + sseg;
  bf16_t* lrow = &Bs[0][srow * 32 + sseg];

  async_ld16(wrow, lrow);
  v8f acc[4] = {};
  const bf16_t* abase = obm + (size_t)(m0 + lm) * DMODEL + (lh << 3);
  int cur = 0;
  #pragma unroll 1
  for (int kk = 0; kk < DMODEL - 32; kk += 32) {
    v16bf a = cat8(*(const v8bf*)(abase + kk), *(const v8bf*)(abase + kk + 16));
    async_ld16(wrow + kk + 32, lrow + (cur ^ 1) * (64 * 32));
    WAIT_ASYNC(1);
    __syncthreads();
    v16bf bfr[4];
    #pragma unroll
    for (int nt = 0; nt < 4; nt++) {
      const bf16_t* bp = &Bs[cur][(nt * 16 + lm) * 32 + (lh << 3)];
      bfr[nt] = cat8(*(const v8bf*)bp, *(const v8bf*)(bp + 16));
    }
    #pragma unroll
    for (int nt = 0; nt < 4; nt++) acc[nt] = wmma_bf16(a, bfr[nt], acc[nt]);
    __syncthreads();
    cur ^= 1;
  }
  {
    const int kk = DMODEL - 32;
    v16bf a = cat8(*(const v8bf*)(abase + kk), *(const v8bf*)(abase + kk + 16));
    WAIT_ASYNC(0);
    __syncthreads();
    v16bf bfr[4];
    #pragma unroll
    for (int nt = 0; nt < 4; nt++) {
      const bf16_t* bp = &Bs[cur][(nt * 16 + lm) * 32 + (lh << 3)];
      bfr[nt] = cat8(*(const v8bf*)bp, *(const v8bf*)(bp + 16));
    }
    #pragma unroll
    for (int nt = 0; nt < 4; nt++) acc[nt] = wmma_bf16(a, bfr[nt], acc[nt]);
  }
  #pragma unroll
  for (int nt = 0; nt < 4; nt++)
    #pragma unroll
    for (int r = 0; r < 8; r++) {
      int i = m0 + r + (lh << 3);
      int j = j0 + nt * 16 + lm;
      out[(size_t)i * DMODEL + j] = acc[nt][r];
    }
}

// ---------------------------------------------------------------------------
// LN2 in-place on fp32 output
// ---------------------------------------------------------------------------
__global__ __launch_bounds__(256) void k_ln2(float* __restrict__ y,
                                             const float* __restrict__ g,
                                             const float* __restrict__ b) {
  __shared__ float s1[256], s2[256];
  int row = blockIdx.x, t = threadIdx.x;
  float* yr = y + (size_t)row * DMODEL;
  float v[4], s = 0.f, ss = 0.f;
  #pragma unroll
  for (int i = 0; i < 4; i++) {
    float u = yr[t + 256 * i];
    v[i] = u; s += u; ss += u * u;
  }
  s1[t] = s; s2[t] = ss; __syncthreads();
  for (int off = 128; off > 0; off >>= 1) {
    if (t < off) { s1[t] += s1[t + off]; s2[t] += s2[t + off]; }
    __syncthreads();
  }
  float mu  = s1[0] * (1.0f / DMODEL);
  float var = s2[0] * (1.0f / DMODEL) - mu * mu;
  float rstd = rsqrtf(var + 1e-5f);
  #pragma unroll
  for (int i = 0; i < 4; i++) {
    int c = t + 256 * i;
    yr[c] = (v[i] - mu) * rstd * g[c] + b[c];
  }
}

// ---------------------------------------------------------------------------
extern "C" void kernel_launch(void* const* d_in, const int* in_sizes, int n_in,
                              void* d_out, int out_size, void* d_ws, size_t ws_size,
                              hipStream_t stream) {
  const float* x     = (const float*)d_in[0];
  const float* ln1_g = (const float*)d_in[1];
  const float* ln1_b = (const float*)d_in[2];
  const float* in_w  = (const float*)d_in[3];
  const float* out_w = (const float*)d_in[4];
  const float* ln2_g = (const float*)d_in[5];
  const float* ln2_b = (const float*)d_in[6];
  const float* slopes = (const float*)d_in[7];
  const float* smear  = (const float*)d_in[8];

  char* ws = (char*)d_ws;
  size_t off = 0;
  auto alloc = [&](size_t bytes) -> char* {
    char* p = ws + off;
    off = (off + bytes + 255) & ~(size_t)255;
    return p;
  };
  bf16_t* hb   = (bf16_t*)alloc((size_t)ROWS * DMODEL * 2);      // LN1 output
  bf16_t* wib  = (bf16_t*)alloc((size_t)QKVCOLS * DMODEL * 2);   // in_w bf16
  bf16_t* wob  = (bf16_t*)alloc((size_t)DMODEL * DMODEL * 2);    // out_w bf16
  bf16_t* qb   = (bf16_t*)alloc((size_t)ROWS * DMODEL * 2);      // Q head-major
  bf16_t* kraw = (bf16_t*)alloc((size_t)ROWS * DMODEL * 2);      // K pre-smear
  bf16_t* kb   = (bf16_t*)alloc((size_t)ROWS * DMODEL * 2);      // K smeared
  bf16_t* vbT  = (bf16_t*)alloc((size_t)ROWS * DMODEL * 2);      // V transposed
  bf16_t* obm  = (bf16_t*)alloc((size_t)ROWS * DMODEL * 2);      // attn output

  k_f2bf<<<(QKVCOLS * DMODEL + 255) / 256, 256, 0, stream>>>(in_w, wib, QKVCOLS * DMODEL);
  k_f2bf<<<(DMODEL * DMODEL + 255) / 256, 256, 0, stream>>>(out_w, wob, DMODEL * DMODEL);
  k_ln1<<<ROWS, 256, 0, stream>>>(x, ln1_g, ln1_b, hb);
  k_qkv<<<32 * 48, 256, 0, stream>>>(hb, wib, qb, kraw, vbT);
  k_smear<<<(ROWS * DMODEL) / 256, 256, 0, stream>>>(kraw, smear, kb);
  k_attn<<<(32 * 128) / 8, 256, 0, stream>>>(qb, kb, vbT, slopes, obm);
  k_proj<<<32 * 16, 256, 0, stream>>>(obm, wob, (float*)d_out);
  k_ln2<<<ROWS, 256, 0, stream>>>((float*)d_out, ln2_g, ln2_b);
}
